// DecoderLayer_39221641347757
// MI455X (gfx1250) — matrix operand order repaired and checked by reading, not compiled, not run on hardware
//
#include <hip/hip_runtime.h>

// ---------------- types ----------------
typedef __attribute__((ext_vector_type(16))) __bf16 v16bf;
typedef __attribute__((ext_vector_type(8)))  float  v8f;
typedef __attribute__((ext_vector_type(4)))  unsigned int v4u;
typedef unsigned short u16;
typedef long long i64;

union Frag { v16bf v; v4u u[2]; };

// fp32 -> bf16, round-to-nearest-even
__device__ __forceinline__ u16 f2bf(float f) {
    unsigned int u = __float_as_uint(f);
    u += 0x7fffu + ((u >> 16) & 1u);
    return (u16)(u >> 16);
}

__device__ __forceinline__ float waveRedSum(float v) {
    #pragma unroll
    for (int o = 16; o > 0; o >>= 1) v += __shfl_xor(v, o, 32);
    return v;
}

// ---- CDNA5 async global->LDS copy (ASYNCcnt-tracked DMA, no VGPR staging) ----
// Low 32 bits of a generic pointer to __shared__ are the LDS byte address
// (ISA 10.2 aperture rules), which is what the VDST operand carries.
__device__ __forceinline__ void async_copy_b128(u16* lds_dst, const u16* g_src) {
    unsigned l = (unsigned)(unsigned long long)lds_dst;
    unsigned long long g = (unsigned long long)g_src;
    asm volatile("global_load_async_to_lds_b128 %0, %1, off"
                 :: "v"(l), "v"(g) : "memory");
}
__device__ __forceinline__ void wait_async_le4() {
    asm volatile("s_wait_asynccnt 4" ::: "memory");
}
__device__ __forceinline__ void wait_async_0() {
    asm volatile("s_wait_asynccnt 0" ::: "memory");
}

// ---------------- problem constants ----------------
#define Bc 2
#define Tc 2048
#define Dc 1024
#define Hc 8
#define Fc 4096
#define DHc 1024          // d_head == d_hidden (source quirk)
#define BT (Bc*Tc)        // 4096

// ---------------- GEMM: C = A * B^T  (bf16 in, f32 acc) ----------------
// A: [M,K] row-major bf16. B: [N,K] row-major bf16 (i.e. B^T applied).
// Workgroup tile 128x128, 8 waves (2 in M x 4 in N), wave tile 64x32,
// K-step 32 via v_wmma_f32_16x16x32_bf16, double-buffered LDS filled by
// global_load_async_to_lds_b128.
#define BM 128
#define BN 128
#define KT 32
#define LDSP 40   // padded LDS K-stride (shorts) to spread banks

enum { EPI_F32 = 0, EPI_BF16 = 1, EPI_RES_F32 = 2, EPI_BIAS_RELU_BF16 = 3,
       EPI_BIAS_RES_F32 = 4, EPI_BF16_VT = 5 };

template<int EPI>
__global__ __launch_bounds__(256) void gemm_bf16_wmma(
    const u16* __restrict__ A, i64 lda, i64 aZb, i64 aZh,
    const u16* __restrict__ Bm, i64 ldb, i64 bZb, i64 bZh,
    void* __restrict__ Cv, i64 ldc, i64 cZb, i64 cZh,
    int M, int N, int Kd, int Hn,
    const float* __restrict__ bias,
    const float* __restrict__ res, i64 ldres,
    int causal)
{
    __shared__ u16 As[2][BM * LDSP];
    __shared__ u16 Bs[2][BN * LDSP];

    const int nBase = blockIdx.x * BN;
    const int mBase = blockIdx.y * BM;
    if (causal && nBase > mBase) return;   // tile fully above causal diagonal

    const int z  = blockIdx.z;
    const int zb = z / Hn, zh = z % Hn;
    const u16* Ab = A  + zb * aZb + zh * aZh;
    const u16* Bb = Bm + zb * bZb + zh * bZh;
    const i64  cOff = (i64)zb * cZb + (i64)zh * cZh;

    const int tid  = threadIdx.x;
    const int lane = tid & 31, wid = tid >> 5;
    const int mW = (wid & 1) * 64;      // wave M offset in tile
    const int nW = (wid >> 1) * 32;     // wave N offset in tile
    const int lh = lane >> 4, lm = lane & 15;

    // each thread stages 2 A-chunks + 2 B-chunks (16B each) per K-step
    const int r0 = tid >> 2,          c0 = (tid & 3) * 8;
    const int r1 = (tid + 256) >> 2,  c1 = c0;   // (idx&3) identical for +256

    auto issueTile = [&](int buf, int kk) {
        async_copy_b128(&As[buf][r0 * LDSP + c0], Ab + (i64)(mBase + r0) * lda + kk + c0);
        async_copy_b128(&Bs[buf][r0 * LDSP + c0], Bb + (i64)(nBase + r0) * ldb + kk + c0);
        async_copy_b128(&As[buf][r1 * LDSP + c1], Ab + (i64)(mBase + r1) * lda + kk + c1);
        async_copy_b128(&Bs[buf][r1 * LDSP + c1], Bb + (i64)(nBase + r1) * ldb + kk + c1);
    };

    v8f acc[4][2];
    v8f zero8 = {};
    #pragma unroll
    for (int im = 0; im < 4; ++im)
        #pragma unroll
        for (int in = 0; in < 2; ++in) acc[im][in] = zero8;

    issueTile(0, 0);                         // prologue: tile 0 in flight

    for (int k0 = 0; k0 < Kd; k0 += KT) {
        const int cur = (k0 / KT) & 1;
        const bool hasNext = (k0 + KT) < Kd;

        __syncthreads();                     // everyone done reading buf[cur^1]
        if (hasNext) {
            issueTile(cur ^ 1, k0 + KT);
            wait_async_le4();                // tile `cur` landed (in-order), next still in flight
        } else {
            wait_async_0();
        }
        __syncthreads();                     // all waves' tile `cur` visible in LDS

        // ---- build fragments (ISA 7.12.2 bf16 layouts) ----
        // A lane: M = lane&15; elems 0..7 -> k = lh*8+0..7, elems 8..15 -> 16+lh*8+0..7
        Frag af[4];
        #pragma unroll
        for (int im = 0; im < 4; ++im) {
            int r = mW + im * 16 + lm;
            af[im].u[0] = *(const v4u*)&As[cur][r * LDSP + lh * 8];
            af[im].u[1] = *(const v4u*)&As[cur][r * LDSP + 16 + lh * 8];
        }
        // B lane: N = lane&15; elems 0..7 -> k = lh*8+0..7, elems 8..15 -> 16+lh*8+0..7
        Frag bf[2];
        #pragma unroll
        for (int in = 0; in < 2; ++in) {
            int r = nW + in * 16 + lm;
            bf[in].u[0] = *(const v4u*)&Bs[cur][r * LDSP + lh * 8];
            bf[in].u[1] = *(const v4u*)&Bs[cur][r * LDSP + 16 + lh * 8];
        }
        #pragma unroll
        for (int im = 0; im < 4; ++im)
            #pragma unroll
            for (int in = 0; in < 2; ++in)
                acc[im][in] = __builtin_amdgcn_wmma_f32_16x16x32_bf16(
                    false, af[im].v, false, bf[in].v, (short)0, acc[im][in], false, false);
    }

    // ---- epilogue: C/D layout: VGPR r -> row r + 8*lh, col = lane&15 ----
    #pragma unroll
    for (int im = 0; im < 4; ++im)
        #pragma unroll
        for (int in = 0; in < 2; ++in)
            #pragma unroll
            for (int r = 0; r < 8; ++r) {
                int row = mBase + mW + im * 16 + lh * 8 + r;
                int col = nBase + nW + in * 16 + lm;
                float vv = acc[im][in][r];
                i64 ci = cOff + (i64)row * ldc + col;
                if constexpr (EPI == EPI_F32) {
                    ((float*)Cv)[ci] = vv;
                } else if constexpr (EPI == EPI_BF16) {
                    ((u16*)Cv)[ci] = f2bf(vv);
                } else if constexpr (EPI == EPI_RES_F32) {
                    ((float*)Cv)[ci] = vv + res[(i64)row * ldres + col];
                } else if constexpr (EPI == EPI_BIAS_RELU_BF16) {
                    float t = vv + bias[col];
                    ((u16*)Cv)[ci] = f2bf(t > 0.f ? t : 0.f);
                } else if constexpr (EPI == EPI_BIAS_RES_F32) {
                    ((float*)Cv)[ci] = vv + bias[col] + res[(i64)row * ldres + col];
                } else {
                    // EPI_BF16_VT: V projection writes V^T laid out [B, H, DH, T]
                    // row = b*Tc + t ; col = head*DHc + dh
                    int bb = row >> 11;           // /Tc
                    int tt = row & (Tc - 1);
                    int hh = col >> 10;           // /DHc
                    int dd = col & (DHc - 1);
                    ((u16*)Cv)[(((i64)(bb * Hc + hh)) * DHc + dd) * Tc + tt] = f2bf(vv);
                }
            }
}

// ---------------- LayerNorm -> bf16 ----------------
__global__ __launch_bounds__(256) void ln_bf16_kernel(
    const float* __restrict__ x, const float* __restrict__ sc,
    const float* __restrict__ bi, u16* __restrict__ y, int D)
{
    const int row = blockIdx.x;
    const float* xr = x + (i64)row * D;
    const int tid = threadIdx.x;
    float s = 0.f, s2 = 0.f;
    for (int i = tid; i < D; i += 256) { float v = xr[i]; s += v; s2 += v * v; }
    s = waveRedSum(s); s2 = waveRedSum(s2);
    __shared__ float red0[8], red1[8], mv[2];
    int lane = tid & 31, wid = tid >> 5;
    if (lane == 0) { red0[wid] = s; red1[wid] = s2; }
    __syncthreads();
    if (tid == 0) {
        float a = 0.f, b = 0.f;
        for (int i = 0; i < 8; ++i) { a += red0[i]; b += red1[i]; }
        float mean = a / D;
        float var  = b / D - mean * mean;
        mv[0] = mean; mv[1] = rsqrtf(var + 1e-5f);
    }
    __syncthreads();
    float mean = mv[0], inv = mv[1];
    u16* yr = y + (i64)row * D;
    for (int i = tid; i < D; i += 256)
        yr[i] = f2bf((xr[i] - mean) * inv * sc[i] + bi[i]);
}

// ---------------- causal softmax (in-place f32 + bf16 copy) ----------------
__global__ __launch_bounds__(256) void softmax_causal_kernel(
    float* __restrict__ attn, u16* __restrict__ attnb, int T, float scale)
{
    const i64 z = blockIdx.x;               // (b,h,q) flat
    const int q = (int)(z % T);
    float* row = attn + z * (i64)T;
    u16*   rb  = attnb + z * (i64)T;
    const int tid = threadIdx.x;
    const int n = q + 1;

    float m = -3.0e38f;
    for (int p = tid; p < n; p += 256) { float v = row[p] * scale; m = v > m ? v : m; }
    #pragma unroll
    for (int o = 16; o > 0; o >>= 1) { float t = __shfl_xor(m, o, 32); m = t > m ? t : m; }
    __shared__ float rmax[8], rsum[8], bro[2];
    int lane = tid & 31, wid = tid >> 5;
    if (lane == 0) rmax[wid] = m;
    __syncthreads();
    if (tid == 0) { float mm = rmax[0]; for (int i = 1; i < 8; ++i) mm = rmax[i] > mm ? rmax[i] : mm; bro[0] = mm; }
    __syncthreads();
    m = bro[0];

    float s = 0.f;
    for (int p = tid; p < n; p += 256) s += __expf(row[p] * scale - m);
    s = waveRedSum(s);
    if (lane == 0) rsum[wid] = s;
    __syncthreads();
    if (tid == 0) { float ss = 0.f; for (int i = 0; i < 8; ++i) ss += rsum[i]; bro[1] = 1.0f / ss; }
    __syncthreads();
    float inv = bro[1];

    for (int p = tid; p < T; p += 256) {
        float v = (p < n) ? __expf(row[p] * scale - m) * inv : 0.0f;
        row[p] = v;
        rb[p] = f2bf(v);
    }
}

// ---------------- fp32 -> bf16 converters ----------------
__global__ __launch_bounds__(256) void cvt_bf16_kernel(
    const float* __restrict__ in, u16* __restrict__ out, i64 n)
{
    for (i64 i = (i64)blockIdx.x * 256 + threadIdx.x; i < n; i += (i64)gridDim.x * 256)
        out[i] = f2bf(in[i]);
}

// W_O [D, DH, H] -> Wo_eff bf16 [D, H*DH] with Wo_eff[d][i*DH+hd] = W_O[d][hd][i]
__global__ __launch_bounds__(256) void permute_wo_kernel(
    const float* __restrict__ in, u16* __restrict__ out)
{
    const i64 n = (i64)Dc * DHc * Hc;
    for (i64 o = (i64)blockIdx.x * 256 + threadIdx.x; o < n; o += (i64)gridDim.x * 256) {
        int hd = (int)(o % DHc);
        i64 t = o / DHc;
        int i = (int)(t % Hc);
        int d = (int)(t / Hc);
        out[o] = f2bf(in[((i64)d * DHc + hd) * Hc + i]);
    }
}

// ---------------- host orchestration ----------------
extern "C" void kernel_launch(void* const* d_in, const int* in_sizes, int n_in,
                              void* d_out, int out_size, void* d_ws, size_t ws_size,
                              hipStream_t stream) {
    (void)in_sizes; (void)n_in; (void)out_size; (void)ws_size;
    const float* x        = (const float*)d_in[0];
    const float* W_K      = (const float*)d_in[1];
    const float* W_Q      = (const float*)d_in[2];
    const float* W_V      = (const float*)d_in[3];
    const float* W_O      = (const float*)d_in[4];
    const float* ln_scale = (const float*)d_in[5];
    const float* ln_bias  = (const float*)d_in[6];
    const float* w1       = (const float*)d_in[7];
    const float* b1       = (const float*)d_in[8];
    const float* w2       = (const float*)d_in[9];
    const float* b2       = (const float*)d_in[10];

    float* out_x = (float*)d_out;
    float* attn  = out_x + (size_t)Bc * Tc * Dc;   // [B,H,T,T] fp32

    // workspace carve-up (256B aligned)
    char* w = (char*)d_ws;
    auto alloc = [&](size_t bytes) -> void* {
        void* p = (void*)w;
        w += (bytes + 255) & ~(size_t)255;
        return p;
    };
    u16*   y_bf   = (u16*)alloc((size_t)BT * Dc * 2);            // LN output (reused for y2)
    u16*   wq_bf  = (u16*)alloc((size_t)Hc * DHc * Dc * 2);
    u16*   wk_bf  = (u16*)alloc((size_t)Hc * DHc * Dc * 2);
    u16*   wv_bf  = (u16*)alloc((size_t)Hc * DHc * Dc * 2);
    u16*   wo_bf  = (u16*)alloc((size_t)Dc * Hc * DHc * 2);
    u16*   w1_bf  = (u16*)alloc((size_t)Fc * Dc * 2);
    u16*   w2_bf  = (u16*)alloc((size_t)Dc * Fc * 2);
    u16*   Qb     = (u16*)alloc((size_t)BT * Hc * DHc * 2);      // [B,T,H,DH]
    u16*   Kb     = (u16*)alloc((size_t)BT * Hc * DHc * 2);      // [B,T,H,DH]
    u16*   Vtb    = (u16*)alloc((size_t)BT * Hc * DHc * 2);      // V^T: [B,H,DH,T]
    u16*   attnb  = (u16*)alloc((size_t)Bc * Hc * Tc * Tc * 2);  // bf16 probs
    u16*   Zb     = (u16*)alloc((size_t)BT * Hc * DHc * 2);      // [B,T,H,DH]
    float* x1     = (float*)alloc((size_t)BT * Dc * 4);          // x + attn residual
    u16*   h_bf   = (u16*)alloc((size_t)BT * Fc * 2);

    const dim3 blk(256);

    // 1) y = LN(x)  -> bf16
    ln_bf16_kernel<<<BT, blk, 0, stream>>>(x, ln_scale, ln_bias, y_bf, Dc);

    // 2) weight downconverts
    cvt_bf16_kernel<<<4096, blk, 0, stream>>>(W_Q, wq_bf, (i64)Hc * DHc * Dc);
    cvt_bf16_kernel<<<4096, blk, 0, stream>>>(W_K, wk_bf, (i64)Hc * DHc * Dc);
    cvt_bf16_kernel<<<4096, blk, 0, stream>>>(W_V, wv_bf, (i64)Hc * DHc * Dc);
    cvt_bf16_kernel<<<4096, blk, 0, stream>>>(w1, w1_bf, (i64)Fc * Dc);
    cvt_bf16_kernel<<<4096, blk, 0, stream>>>(w2, w2_bf, (i64)Dc * Fc);
    permute_wo_kernel<<<4096, blk, 0, stream>>>(W_O, wo_bf);

    // 3) Q/K/V projections: [BT,1024] @ [8192,1024]^T
    {
        dim3 grid(Hc * DHc / BN, BT / BM, 1);
        gemm_bf16_wmma<EPI_BF16><<<grid, blk, 0, stream>>>(
            y_bf, Dc, 0, 0, wq_bf, Dc, 0, 0, Qb, (i64)Hc * DHc, 0, 0,
            BT, Hc * DHc, Dc, 1, nullptr, nullptr, 0, 0);
        gemm_bf16_wmma<EPI_BF16><<<grid, blk, 0, stream>>>(
            y_bf, Dc, 0, 0, wk_bf, Dc, 0, 0, Kb, (i64)Hc * DHc, 0, 0,
            BT, Hc * DHc, Dc, 1, nullptr, nullptr, 0, 0);
        // V written transposed -> [B,H,DH,T] so attn@V is a plain A*B^T GEMM
        gemm_bf16_wmma<EPI_BF16_VT><<<grid, blk, 0, stream>>>(
            y_bf, Dc, 0, 0, wv_bf, Dc, 0, 0, Vtb, 0, 0, 0,
            BT, Hc * DHc, Dc, 1, nullptr, nullptr, 0, 0);
    }

    // 4) scores[b,h] = Q(b,h) @ K(b,h)^T  -> fp32 into d_out attn region (causal skip)
    {
        dim3 grid(Tc / BN, Tc / BM, Bc * Hc);
        gemm_bf16_wmma<EPI_F32><<<grid, blk, 0, stream>>>(
            Qb, (i64)Hc * DHc, (i64)Tc * Hc * DHc, DHc,
            Kb, (i64)Hc * DHc, (i64)Tc * Hc * DHc, DHc,
            attn, Tc, (i64)Hc * Tc * Tc, (i64)Tc * Tc,
            Tc, Tc, DHc, Hc, nullptr, nullptr, 0, /*causal=*/1);
    }

    // 5) causal softmax: fp32 probs in place (d_out) + bf16 copy for next GEMM
    softmax_causal_kernel<<<Bc * Hc * Tc, blk, 0, stream>>>(
        attn, attnb, Tc, 0.03125f /* 1/sqrt(1024) */);

    // 6) Z(b,h) = attn(b,h) [T,T] @ (V^T(b,h) [DH,T])^T -> [B,T,H,DH] bf16
    {
        dim3 grid(DHc / BN, Tc / BM, Bc * Hc);
        gemm_bf16_wmma<EPI_BF16><<<grid, blk, 0, stream>>>(
            attnb, Tc, (i64)Hc * Tc * Tc, (i64)Tc * Tc,
            Vtb, Tc, (i64)Hc * DHc * Tc, (i64)DHc * Tc,
            Zb, (i64)Hc * DHc, (i64)Tc * Hc * DHc, DHc,
            Tc, DHc, Tc, Hc, nullptr, nullptr, 0, 0);
    }

    // 7) x1 = x + Z_all @ Wo_eff^T   ([BT,8192] @ [1024,8192]^T)
    {
        dim3 grid(Dc / BN, BT / BM, 1);
        gemm_bf16_wmma<EPI_RES_F32><<<grid, blk, 0, stream>>>(
            Zb, (i64)Hc * DHc, 0, 0, wo_bf, (i64)Hc * DHc, 0, 0,
            x1, Dc, 0, 0, BT, Dc, Hc * DHc, 1, nullptr, x, Dc, 0);
    }

    // 8) y2 = LN(x1) (same scale/bias — source bug preserved), reuse y_bf
    ln_bf16_kernel<<<BT, blk, 0, stream>>>(x1, ln_scale, ln_bias, y_bf, Dc);

    // 9) h = relu(y2 @ w1^T + b1) -> bf16
    {
        dim3 grid(Fc / BN, BT / BM, 1);
        gemm_bf16_wmma<EPI_BIAS_RELU_BF16><<<grid, blk, 0, stream>>>(
            y_bf, Dc, 0, 0, w1_bf, Dc, 0, 0, h_bf, Fc, 0, 0,
            BT, Fc, Dc, 1, b1, nullptr, 0, 0);
    }

    // 10) out = x1 + h @ w2^T + b2 -> fp32 d_out
    {
        dim3 grid(Dc / BN, BT / BM, 1);
        gemm_bf16_wmma<EPI_BIAS_RES_F32><<<grid, blk, 0, stream>>>(
            h_bf, Fc, 0, 0, w2_bf, Fc, 0, 0, out_x, Dc, 0, 0,
            BT, Dc, Fc, 1, b2, x1, Dc, 0);
    }
}
